// FiLMMambaBlock_41171556499480
// MI455X (gfx1250) — matrix operand order repaired and checked by reading, not compiled, not run on hardware
//
#include <hip/hip_runtime.h>

typedef unsigned short u16;
typedef unsigned int   u32;
typedef __bf16 v16bf __attribute__((ext_vector_type(16)));
typedef float  v8f   __attribute__((ext_vector_type(8)));
typedef u32    u32x4 __attribute__((ext_vector_type(4)));
typedef u32    u32x2 __attribute__((ext_vector_type(2)));
typedef float  f32x4 __attribute__((ext_vector_type(4)));

#define S_LEN 4096
#define D_DIM 1024
#define C_DIM 512
#define M_TOT 16384   // B*S
#define LDP   40      // LDS row stride (bf16 elems): 80B rows, 16B-aligned frag reads

// ---------- bf16 helpers (RNE) ----------
__device__ inline u16 f2bf(float f) {
    u32 u = __float_as_uint(f);
    u32 r = (u + 0x7FFFu + ((u >> 16) & 1u)) >> 16;
    return (u16)r;
}
__device__ inline float bf2f(u16 h) { return __uint_as_float(((u32)h) << 16); }
__device__ inline u32 pack2(float a, float b) {
    return (u32)f2bf(a) | ((u32)f2bf(b) << 16);
}

// ---------- async global -> LDS (CDNA5, ASYNCcnt-tracked) ----------
__device__ inline unsigned lds_off_of(const void* p) {
    return (unsigned)(size_t)(__attribute__((address_space(3))) const void*)p;
}
__device__ inline void async_g2l_b128(unsigned lds_off, const void* gptr) {
    asm volatile("global_load_async_to_lds_b128 %0, %1, off"
                 :: "v"(lds_off), "v"(gptr)
                 : "memory");
}
#define WAIT_ASYNC_0() asm volatile("s_wait_asynccnt 0" ::: "memory")
#define WAIT_ASYNC_4() asm volatile("s_wait_asynccnt 4" ::: "memory")
#define WAIT_ASYNC_6() asm volatile("s_wait_asynccnt 6" ::: "memory")

// ---------- WMMA fragment plumbing ----------
union FragU { v16bf v; u32x4 q[2]; };

// A-matrix 16x32 bf16 (ISA 7.12.2): lane L -> m=L&15, h=L>>4.
// elements 0..7  <- K = 8h .. 8h+7 ; elements 8..15 <- K = 16+8h .. 16+8h+7
__device__ inline v16bf load_fragA(const u16* s, int row0, int lane) {
    const int r = row0 + (lane & 15);
    const int h = (lane >> 4) & 1;
    FragU f;
    f.q[0] = *(const u32x4*)(s + r * LDP + 8 * h);
    f.q[1] = *(const u32x4*)(s + r * LDP + 16 + 8 * h);
    return f.v;
}
// B-matrix 32x16 bf16 (N across lanes; n=L&15, h=L>>4; same K order as A)
// stored N-major in LDS: elements 0..7 <- K = 8h .. 8h+7 ; elements 8..15 <- K = 16+8h .. 16+8h+7
__device__ inline v16bf load_fragB(const u16* s, int row0, int lane) {
    const int r = row0 + (lane & 15);
    const int h = (lane >> 4) & 1;
    FragU f;
    f.q[0] = *(const u32x4*)(s + r * LDP + 8 * h);
    f.q[1] = *(const u32x4*)(s + r * LDP + 16 + 8 * h);
    return f.v;
}
__device__ inline v8f wmma_bf16(v16bf a, v16bf b, v8f c) {
    return __builtin_amdgcn_wmma_f32_16x16x32_bf16(false, a, false, b, (short)0, c, false, false);
}

// ---------- transpose + fp32->bf16 convert: src[K][N] -> dst[N][K] ----------
__global__ __launch_bounds__(256) void transpose_bf16_kernel(
    const float* __restrict__ src, u16* __restrict__ dst, int K, int N) {
    __shared__ float tile[32][33];
    const int nb = blockIdx.x * 32, kb = blockIdx.y * 32;
    const int tx = threadIdx.x & 31, tg = threadIdx.x >> 5;  // tg: 0..7
#pragma unroll
    for (int i = 0; i < 4; ++i) {
        const int kr = tg * 4 + i;
        tile[kr][tx] = src[(size_t)(kb + kr) * N + nb + tx];
    }
    __syncthreads();
#pragma unroll
    for (int i = 0; i < 4; ++i) {
        const int nr = tg * 4 + i;
        dst[(size_t)(nb + nr) * K + kb + tx] = f2bf(tile[tx][nr]);
    }
}

// ---------- FiLM: scale = sigmoid(cond@scale_w + b), shift = cond@shift_w + b ----------
__global__ __launch_bounds__(256) void film_kernel(
    const float* __restrict__ cond,
    const float* __restrict__ sw, const float* __restrict__ sb,
    const float* __restrict__ hw, const float* __restrict__ hb,
    float* __restrict__ scale, float* __restrict__ shift) {
    const int b = blockIdx.x;
    const int d = blockIdx.y * 256 + threadIdx.x;
    float as = 0.f, ah = 0.f;
    for (int c = 0; c < C_DIM; ++c) {
        const float cv = cond[b * C_DIM + c];
        as += cv * sw[c * D_DIM + d];
        ah += cv * hw[c * D_DIM + d];
    }
    const float sv = as + sb[d];
    scale[b * D_DIM + d] = 1.f / (1.f + __expf(-sv));
    shift[b * D_DIM + d] = ah + hb[d];
}

// ---------- LayerNorm over D=1024, out bf16 ----------
__global__ __launch_bounds__(256) void ln_kernel(
    const float* __restrict__ x, const float* __restrict__ g,
    const float* __restrict__ b, u16* __restrict__ out) {
    const int row = blockIdx.x, tid = threadIdx.x;
    const f32x4 v = ((const f32x4*)(x + (size_t)row * D_DIM))[tid];
    float s  = v.x + v.y + v.z + v.w;
    float s2 = v.x * v.x + v.y * v.y + v.z * v.z + v.w * v.w;
#pragma unroll
    for (int off = 16; off > 0; off >>= 1) {
        s  += __shfl_down(s, off, 32);
        s2 += __shfl_down(s2, off, 32);
    }
    __shared__ float rs[8], rs2[8], mv[2];
    const int lane = tid & 31, wid = tid >> 5;
    if (lane == 0) { rs[wid] = s; rs2[wid] = s2; }
    __syncthreads();
    if (tid == 0) {
        float a = 0.f, c = 0.f;
#pragma unroll
        for (int i = 0; i < 8; ++i) { a += rs[i]; c += rs2[i]; }
        const float mean = a * (1.f / 1024.f);
        const float var  = c * (1.f / 1024.f) - mean * mean;
        mv[0] = mean;
        mv[1] = rsqrtf(var + 1e-5f);
    }
    __syncthreads();
    const float mean = mv[0], rstd = mv[1];
    const f32x4 gg = ((const f32x4*)g)[tid];
    const f32x4 bb = ((const f32x4*)b)[tid];
    const float o0 = (v.x - mean) * rstd * gg.x + bb.x;
    const float o1 = (v.y - mean) * rstd * gg.y + bb.y;
    const float o2 = (v.z - mean) * rstd * gg.z + bb.z;
    const float o3 = (v.w - mean) * rstd * gg.w + bb.w;
    u32x2 pk; pk.x = pack2(o0, o1); pk.y = pack2(o2, o3);
    ((u32x2*)(out + (size_t)row * D_DIM))[tid] = pk;
}

// ---------- depthwise conv K=5, same padding (per-batch along S), bf16 in/out ----------
__global__ __launch_bounds__(256) void dwconv_kernel(
    const u16* __restrict__ nin, const float* __restrict__ w,
    const float* __restrict__ bias, u16* __restrict__ out) {
    const int m = blockIdx.x;       // b*S + s
    const int s = m & (S_LEN - 1);
    const int d0 = threadIdx.x * 4;
    float a0 = bias[d0], a1 = bias[d0 + 1], a2 = bias[d0 + 2], a3 = bias[d0 + 3];
#pragma unroll
    for (int k = 0; k < 5; ++k) {
        const int sp = s + k - 2;
        if (sp < 0 || sp >= S_LEN) continue;
        const u32x2 q = *(const u32x2*)(nin + (size_t)(m + k - 2) * D_DIM + d0);
        a0 += bf2f((u16)(q.x & 0xffff)) * w[(d0 + 0) * 5 + k];
        a1 += bf2f((u16)(q.x >> 16))    * w[(d0 + 1) * 5 + k];
        a2 += bf2f((u16)(q.y & 0xffff)) * w[(d0 + 2) * 5 + k];
        a3 += bf2f((u16)(q.y >> 16))    * w[(d0 + 3) * 5 + k];
    }
    u32x2 pk; pk.x = pack2(a0, a1); pk.y = pack2(a2, a3);
    ((u32x2*)(out + (size_t)m * D_DIM))[threadIdx.x] = pk;
}

// ---------- GEMM1: u = sigmoid(conv@Wg + bg) * (conv@Wv + bv), bf16 out ----------
// workgroup tile: 128 M x 128 N(d), waves 4x2, wave tile 32x64 per half, K-step 32
// double-buffered LDS filled with async global->LDS copies (ASYNCcnt pipeline)
__global__ __launch_bounds__(256) void gemm1_kernel(
    const u16* __restrict__ conv, const u16* __restrict__ gwT,
    const float* __restrict__ gate_b, u16* __restrict__ u) {
    __shared__ __align__(16) u16 As[2][128 * LDP];
    __shared__ __align__(16) u16 Bg[2][128 * LDP];
    __shared__ __align__(16) u16 Bv[2][128 * LDP];
    const int tid = threadIdx.x, lane = tid & 31, wid = tid >> 5;
    const int wm = wid & 3, wn = wid >> 2;
    const int mblk = blockIdx.x * 128, nblk = blockIdx.y * 128;
    const int m0w = wm * 32, n0w = wn * 64;
    const int frow = tid >> 1, fcol = (tid & 1) * 16;

    const u16* gA = conv + (size_t)(mblk + frow) * D_DIM + fcol;
    const u16* gB = gwT + (size_t)(nblk + frow) * D_DIM + fcol;
    const u16* gV = gwT + (size_t)(D_DIM + nblk + frow) * D_DIM + fcol;
    unsigned lA[2], lB[2], lV[2];
#pragma unroll
    for (int i = 0; i < 2; ++i) {
        lA[i] = lds_off_of(&As[i][frow * LDP + fcol]);
        lB[i] = lds_off_of(&Bg[i][frow * LDP + fcol]);
        lV[i] = lds_off_of(&Bv[i][frow * LDP + fcol]);
    }

    const v8f z = {0.f, 0.f, 0.f, 0.f, 0.f, 0.f, 0.f, 0.f};
    v8f accg[2][4], accv[2][4];
#pragma unroll
    for (int i = 0; i < 2; ++i)
#pragma unroll
        for (int j = 0; j < 4; ++j) { accg[i][j] = z; accv[i][j] = z; }

    // prologue: buffer 0 <- k0 = 0 (6 async b128 per thread)
    async_g2l_b128(lA[0],      gA);
    async_g2l_b128(lA[0] + 16, gA + 8);
    async_g2l_b128(lB[0],      gB);
    async_g2l_b128(lB[0] + 16, gB + 8);
    async_g2l_b128(lV[0],      gV);
    async_g2l_b128(lV[0] + 16, gV + 8);

    for (int it = 0; it < D_DIM / 32; ++it) {
        const int cur = it & 1;
        if (it + 1 < D_DIM / 32) {
            const int k1 = (it + 1) * 32;
            const int nb = cur ^ 1;
            async_g2l_b128(lA[nb],      gA + k1);
            async_g2l_b128(lA[nb] + 16, gA + k1 + 8);
            async_g2l_b128(lB[nb],      gB + k1);
            async_g2l_b128(lB[nb] + 16, gB + k1 + 8);
            async_g2l_b128(lV[nb],      gV + k1);
            async_g2l_b128(lV[nb] + 16, gV + k1 + 8);
            WAIT_ASYNC_6();   // in-order completion => current buffer resident
        } else {
            WAIT_ASYNC_0();
        }
        __syncthreads();
        const v16bf a0 = load_fragA(As[cur], m0w, lane);
        const v16bf a1 = load_fragA(As[cur], m0w + 16, lane);
#pragma unroll
        for (int nt = 0; nt < 4; ++nt) {
            const v16bf bg = load_fragB(Bg[cur], n0w + nt * 16, lane);
            accg[0][nt] = wmma_bf16(a0, bg, accg[0][nt]);
            accg[1][nt] = wmma_bf16(a1, bg, accg[1][nt]);
            const v16bf bv = load_fragB(Bv[cur], n0w + nt * 16, lane);
            accv[0][nt] = wmma_bf16(a0, bv, accv[0][nt]);
            accv[1][nt] = wmma_bf16(a1, bv, accv[1][nt]);
        }
        __syncthreads();  // all waves done reading buf cur before it is refilled
    }
    // epilogue: u = sigmoid(g+bg) * (v+bv)
    const int col = lane & 15, half = lane >> 4;
#pragma unroll
    for (int mt = 0; mt < 2; ++mt)
#pragma unroll
        for (int nt = 0; nt < 4; ++nt) {
            const int n = nblk + n0w + nt * 16 + col;
            const float bg0 = gate_b[n];
            const float bv0 = gate_b[D_DIM + n];
#pragma unroll
            for (int r = 0; r < 8; ++r) {
                const int m = mblk + m0w + mt * 16 + r + 8 * half;
                const float g  = accg[mt][nt][r] + bg0;
                const float vv = accv[mt][nt][r] + bv0;
                const float ug = vv / (1.f + __expf(-g));
                u[(size_t)m * D_DIM + n] = f2bf(ug);
            }
        }
}

// ---------- GEMM2: out = x + (u@Wp + bp)*scale[b] + shift[b] ----------
__global__ __launch_bounds__(256) void gemm2_kernel(
    const u16* __restrict__ uin, const u16* __restrict__ pwT,
    const float* __restrict__ proj_b, const float* __restrict__ x,
    const float* __restrict__ scale, const float* __restrict__ shift,
    float* __restrict__ out) {
    __shared__ __align__(16) u16 As[2][128 * LDP];
    __shared__ __align__(16) u16 Bs[2][128 * LDP];
    const int tid = threadIdx.x, lane = tid & 31, wid = tid >> 5;
    const int wm = wid & 3, wn = wid >> 2;
    const int mblk = blockIdx.x * 128, nblk = blockIdx.y * 128;
    const int m0w = wm * 32, n0w = wn * 64;
    const int frow = tid >> 1, fcol = (tid & 1) * 16;

    const u16* gA = uin + (size_t)(mblk + frow) * D_DIM + fcol;
    const u16* gB = pwT + (size_t)(nblk + frow) * D_DIM + fcol;
    unsigned lA[2], lB[2];
#pragma unroll
    for (int i = 0; i < 2; ++i) {
        lA[i] = lds_off_of(&As[i][frow * LDP + fcol]);
        lB[i] = lds_off_of(&Bs[i][frow * LDP + fcol]);
    }

    const v8f z = {0.f, 0.f, 0.f, 0.f, 0.f, 0.f, 0.f, 0.f};
    v8f acc[2][4];
#pragma unroll
    for (int i = 0; i < 2; ++i)
#pragma unroll
        for (int j = 0; j < 4; ++j) acc[i][j] = z;

    async_g2l_b128(lA[0],      gA);
    async_g2l_b128(lA[0] + 16, gA + 8);
    async_g2l_b128(lB[0],      gB);
    async_g2l_b128(lB[0] + 16, gB + 8);

    for (int it = 0; it < D_DIM / 32; ++it) {
        const int cur = it & 1;
        if (it + 1 < D_DIM / 32) {
            const int k1 = (it + 1) * 32;
            const int nb = cur ^ 1;
            async_g2l_b128(lA[nb],      gA + k1);
            async_g2l_b128(lA[nb] + 16, gA + k1 + 8);
            async_g2l_b128(lB[nb],      gB + k1);
            async_g2l_b128(lB[nb] + 16, gB + k1 + 8);
            WAIT_ASYNC_4();
        } else {
            WAIT_ASYNC_0();
        }
        __syncthreads();
        const v16bf a0 = load_fragA(As[cur], m0w, lane);
        const v16bf a1 = load_fragA(As[cur], m0w + 16, lane);
#pragma unroll
        for (int nt = 0; nt < 4; ++nt) {
            const v16bf bb = load_fragB(Bs[cur], n0w + nt * 16, lane);
            acc[0][nt] = wmma_bf16(a0, bb, acc[0][nt]);
            acc[1][nt] = wmma_bf16(a1, bb, acc[1][nt]);
        }
        __syncthreads();
    }
    const int col = lane & 15, half = lane >> 4;
#pragma unroll
    for (int mt = 0; mt < 2; ++mt)
#pragma unroll
        for (int nt = 0; nt < 4; ++nt) {
            const int n = nblk + n0w + nt * 16 + col;
            const float pb = proj_b[n];
#pragma unroll
            for (int r = 0; r < 8; ++r) {
                const int m = mblk + m0w + mt * 16 + r + 8 * half;
                const int b = m >> 12;  // S = 4096
                const float h = acc[mt][nt][r] + pb;
                const size_t idx = (size_t)m * D_DIM + n;
                out[idx] = x[idx] + h * scale[b * D_DIM + n] + shift[b * D_DIM + n];
            }
        }
}

extern "C" void kernel_launch(void* const* d_in, const int* in_sizes, int n_in,
                              void* d_out, int out_size, void* d_ws, size_t ws_size,
                              hipStream_t stream) {
    const float* x       = (const float*)d_in[0];
    const float* cond    = (const float*)d_in[1];
    const float* ln_g    = (const float*)d_in[2];
    const float* ln_b    = (const float*)d_in[3];
    const float* dw_w    = (const float*)d_in[4];
    const float* dw_b    = (const float*)d_in[5];
    const float* gate_w  = (const float*)d_in[6];
    const float* gate_b  = (const float*)d_in[7];
    const float* proj_w  = (const float*)d_in[8];
    const float* proj_b  = (const float*)d_in[9];
    const float* scale_w = (const float*)d_in[10];
    const float* scale_b = (const float*)d_in[11];
    const float* shift_w = (const float*)d_in[12];
    const float* shift_b = (const float*)d_in[13];
    float* out = (float*)d_out;

    char* ws = (char*)d_ws;
    u16*   nbuf  = (u16*)(ws + 0);          // 32MB: LN output, later reused for u
    u16*   convb = (u16*)(ws + 33554432);   // 32MB
    u16*   gwT   = (u16*)(ws + 67108864);   // 4MB  [2048][1024] bf16
    u16*   pwT   = (u16*)(ws + 71303168);   // 2MB  [1024][1024] bf16
    float* scl   = (float*)(ws + 73400320); // 16KB [4][1024]
    float* shf   = (float*)(ws + 73416704); // 16KB [4][1024]

    transpose_bf16_kernel<<<dim3(64, 32), 256, 0, stream>>>(gate_w, gwT, 1024, 2048);
    transpose_bf16_kernel<<<dim3(32, 32), 256, 0, stream>>>(proj_w, pwT, 1024, 1024);
    film_kernel<<<dim3(4, 4), 256, 0, stream>>>(cond, scale_w, scale_b, shift_w, shift_b, scl, shf);
    ln_kernel<<<M_TOT, 256, 0, stream>>>(x, ln_g, ln_b, nbuf);
    dwconv_kernel<<<M_TOT, 256, 0, stream>>>(nbuf, dw_w, dw_b, convb);
    gemm1_kernel<<<dim3(128, 8), 256, 0, stream>>>(convb, gwT, gate_b, nbuf /* u */);
    gemm2_kernel<<<dim3(128, 8), 256, 0, stream>>>(nbuf, pwT, proj_b, x, scl, shf, out);
}
